// RWKV7_StateInfer_64647847739565
// MI455X (gfx1250) — matrix operand order repaired and checked by reading, not compiled, not run on hardware
//
#include <hip/hip_runtime.h>
#include <hip/hip_bf16.h>
#include <math.h>

typedef __attribute__((ext_vector_type(16))) __bf16 v16bf;
typedef __attribute__((ext_vector_type(8)))  float  v8f;
typedef int v4i __attribute__((vector_size(16)));

constexpr int Bb = 2, Tt = 1024, Cc = 1024, Ll = 2, Hh = 16, Nn = 64, FFf = 4096, Vv = 32000;
constexpr int BT = Bb * Tt;

#if __has_builtin(__builtin_amdgcn_global_load_async_to_lds_b128)
#define HAS_ASYNC 1
#else
#define HAS_ASYNC 0
#endif

// 16-byte global->LDS copy: CDNA5 async DMA when available (ASYNCcnt-tracked),
// otherwise VGPR round-trip (ds_store).
__device__ __forceinline__ void async_cp16(const void* g, void* l) {
#if HAS_ASYNC
  __builtin_amdgcn_global_load_async_to_lds_b128(
      (__attribute__((address_space(1))) v4i*)(g),
      (__attribute__((address_space(3))) v4i*)(l), 0, 0);
#else
  *(uint4*)l = *(const uint4*)g;
#endif
}

__device__ __forceinline__ void wait_async0() {
#if HAS_ASYNC
#if __has_builtin(__builtin_amdgcn_s_wait_asynccnt)
  __builtin_amdgcn_s_wait_asynccnt(0);
#else
  asm volatile("s_wait_asynccnt 0x0" ::: "memory");
#endif
#endif
}

// ---------------------------------------------------------------------------
// WMMA GEMM: Out[M,N] = epilogue(A[M,K] @ W[K,N]).
// A: fp32 (converted to bf16 while staging); W: pre-converted bf16, TRANSPOSED
// [N][K] so B-tiles are contiguous per output column -> async-copy friendly.
// Tile 128x128x32, 256 threads (8 waves), wave = 32x64 sub-tile = 2x4 WMMAs.
// Double-buffered LDS, one barrier per K-step, async B staging.
// EPI: 0=store 1=tanh 2=sigmoid 3=relu^2 4=+aux(resid) 5=sigmoid(aux[col]+v)
// ---------------------------------------------------------------------------
constexpr int BM = 128, BN = 128, BK = 32;

template <int EPI>
__global__ __launch_bounds__(256) void gemm_wmma(
    const float* __restrict__ A, const __bf16* __restrict__ Wt,
    float* Out, const float* aux, int M, int N, int K) {
  __shared__ alignas(128) __bf16 As[2][BM][BK];       // 2 x 8 KB
  __shared__ alignas(128) __bf16 Bs[2][BN][BK + 8];   // 2 x 10 KB (col-major tile)
  const int tid  = threadIdx.x;
  const int m0   = blockIdx.y * BM;
  const int n0   = blockIdx.x * BN;
  const int lane = tid & 31;
  const int wave = tid >> 5;
  const int wm   = wave & 3;     // 32-row slab
  const int wn   = wave >> 2;    // 64-col slab
  const int khalf = lane >> 4;   // ISA 16-bit layout K-half
  const int l16   = lane & 15;

  v8f acc[2][4];
#pragma unroll
  for (int mi = 0; mi < 2; ++mi)
#pragma unroll
    for (int ni = 0; ni < 4; ++ni)
#pragma unroll
      for (int v = 0; v < 8; ++v) acc[mi][ni][v] = 0.f;

  float4 areg[4];
  const int arow = tid >> 3;          // 0..31 (x4 passes)
  const int ac4  = (tid & 7) * 4;     // 0..28

  auto loadA = [&](int k0) {
#pragma unroll
    for (int p = 0; p < 4; ++p)
      areg[p] = *(const float4*)(A + (size_t)(m0 + p * 32 + arow) * K + k0 + ac4);
  };
  auto storeA = [&](int buf) {
#pragma unroll
    for (int p = 0; p < 4; ++p) {
      const int row = p * 32 + arow;
      As[buf][row][ac4 + 0] = (__bf16)areg[p].x;
      As[buf][row][ac4 + 1] = (__bf16)areg[p].y;
      As[buf][row][ac4 + 2] = (__bf16)areg[p].z;
      As[buf][row][ac4 + 3] = (__bf16)areg[p].w;
    }
  };
  auto issueB = [&](int buf, int k0) {
#pragma unroll
    for (int q = 0; q < 2; ++q) {
      const int idx  = tid * 2 + q;     // 512 16B chunks = 8 KB tile
      const int col  = idx >> 2;        // 0..127
      const int quad = idx & 3;         // 8 bf16 per chunk
      int nc = n0 + col; if (nc > N - 1) nc = N - 1;   // clamp: dead cols never stored
      async_cp16(Wt + (size_t)nc * K + k0 + quad * 8, &Bs[buf][col][quad * 8]);
    }
  };

  const int Ksteps = K / BK;
  issueB(0, 0);
  loadA(0);
  storeA(0);
  int buf = 0;
  for (int kt = 0; kt < Ksteps; ++kt) {
    wait_async0();        // current B tile resident in LDS
    __syncthreads();      // A stores visible; all waves done with buf^1
    const bool more = (kt + 1 < Ksteps);
    if (more) {
      issueB(buf ^ 1, (kt + 1) * BK);   // overlap with this step's WMMAs
      loadA((kt + 1) * BK);
      __builtin_prefetch(A + (size_t)(m0 + arow) * K + (kt + 2) * BK, 0, 1);
    }

    v16bf afr[2], bfr[4];
#pragma unroll
    for (int mi = 0; mi < 2; ++mi) {
      const int row = wm * 32 + mi * 16 + l16;
#pragma unroll
      for (int j = 0; j < 16; ++j) {
        const int kk = khalf * 16 + j;  // contiguous K order, matches B
        afr[mi][j] = As[buf][row][kk];
      }
    }
#pragma unroll
    for (int ni = 0; ni < 4; ++ni) {
      const int col = wn * 64 + ni * 16 + l16;
#pragma unroll
      for (int j = 0; j < 16; ++j)
        bfr[ni][j] = Bs[buf][col][khalf * 16 + j];                // contiguous K order, matches A
    }
#pragma unroll
    for (int mi = 0; mi < 2; ++mi)
#pragma unroll
      for (int ni = 0; ni < 4; ++ni)
        acc[mi][ni] = __builtin_amdgcn_wmma_f32_16x16x32_bf16(
            false, afr[mi], false, bfr[ni], (short)0, acc[mi][ni], false, false);

    if (more) storeA(buf ^ 1);
    buf ^= 1;
  }

  // Epilogue. C/D layout: VGPR v -> row (lane>>4)*8 + v, col = lane&15.
#pragma unroll
  for (int mi = 0; mi < 2; ++mi)
#pragma unroll
    for (int ni = 0; ni < 4; ++ni) {
      const int col = n0 + wn * 64 + ni * 16 + l16;
      if (col >= N) continue;
#pragma unroll
      for (int v = 0; v < 8; ++v) {
        const int row = m0 + wm * 32 + mi * 16 + khalf * 8 + v;
        const size_t o = (size_t)row * N + col;
        float val = acc[mi][ni][v];
        if (EPI == 1) val = tanhf(val);
        else if (EPI == 2) val = 1.f / (1.f + expf(-val));
        else if (EPI == 3) { const float t = fmaxf(val, 0.f); val = t * t; }
        else if (EPI == 4) val += aux[o];
        else if (EPI == 5) val = 1.f / (1.f + expf(-(aux[col] + val)));
        Out[o] = val;
      }
    }
}

static inline void gemm(const float* A, const __bf16* Wt, float* O, const float* aux,
                        int M, int N, int K, int epi, hipStream_t s) {
  dim3 g((N + BN - 1) / BN, M / BM), b(256);
  switch (epi) {
    case 0: gemm_wmma<0><<<g, b, 0, s>>>(A, Wt, O, aux, M, N, K); break;
    case 1: gemm_wmma<1><<<g, b, 0, s>>>(A, Wt, O, aux, M, N, K); break;
    case 2: gemm_wmma<2><<<g, b, 0, s>>>(A, Wt, O, aux, M, N, K); break;
    case 3: gemm_wmma<3><<<g, b, 0, s>>>(A, Wt, O, aux, M, N, K); break;
    case 4: gemm_wmma<4><<<g, b, 0, s>>>(A, Wt, O, aux, M, N, K); break;
    case 5: gemm_wmma<5><<<g, b, 0, s>>>(A, Wt, O, aux, M, N, K); break;
  }
}

// fp32 [K][N] -> bf16 transposed [N][K]  (tiled transpose through LDS)
__global__ __launch_bounds__(256) void k_cvt_t(
    const float* __restrict__ src, __bf16* __restrict__ dst, int K, int N) {
  __shared__ float tile[32][33];
  const int kb = blockIdx.y * 32, nb = blockIdx.x * 32;
  const int tx = threadIdx.x & 31, ty = threadIdx.x >> 5;  // 8 rows/pass
#pragma unroll
  for (int p = 0; p < 32; p += 8) {
    const int k = kb + ty + p, n = nb + tx;
    tile[ty + p][tx] = (k < K && n < N) ? src[(size_t)k * N + n] : 0.f;
  }
  __syncthreads();
#pragma unroll
  for (int p = 0; p < 32; p += 8) {
    const int n = nb + ty + p, k = kb + tx;
    if (n < N && k < K) dst[(size_t)n * K + k] = (__bf16)tile[tx][ty + p];
  }
}

// ---------------------------------------------------------------------------
// Elementwise / normalization kernels
// ---------------------------------------------------------------------------
__device__ __forceinline__ float bsum(float v, float* red) {
  const int t = threadIdx.x;
  red[t] = v; __syncthreads();
  for (int s = 128; s > 0; s >>= 1) { if (t < s) red[t] += red[t + s]; __syncthreads(); }
  const float r = red[0]; __syncthreads();
  return r;
}

__global__ __launch_bounds__(256) void k_embed_ln0(
    const int* __restrict__ idx, const float* __restrict__ emb,
    const float* __restrict__ g, const float* __restrict__ b, float* __restrict__ x) {
  __shared__ float red[256];
  const int tok = blockIdx.x, tid = threadIdx.x;
  const float* e = emb + (size_t)idx[tok] * Cc;
  float v[4]; float s = 0.f;
#pragma unroll
  for (int u = 0; u < 4; ++u) { v[u] = e[tid * 4 + u]; s += v[u]; }
  const float mean = bsum(s, red) * (1.f / Cc);
  float q = 0.f;
#pragma unroll
  for (int u = 0; u < 4; ++u) { const float d = v[u] - mean; q += d * d; }
  const float rs = rsqrtf(bsum(q, red) * (1.f / Cc) + 1e-5f);
#pragma unroll
  for (int u = 0; u < 4; ++u) {
    const int c = tid * 4 + u;
    x[(size_t)tok * Cc + c] = (v[u] - mean) * rs * g[c] + b[c];
  }
}

__global__ __launch_bounds__(256) void k_ln(
    const float* __restrict__ x, const float* __restrict__ g,
    const float* __restrict__ b, float* __restrict__ out) {
  __shared__ float red[256];
  const int tok = blockIdx.x, tid = threadIdx.x;
  const float* e = x + (size_t)tok * Cc;
  float v[4]; float s = 0.f;
#pragma unroll
  for (int u = 0; u < 4; ++u) { v[u] = e[tid * 4 + u]; s += v[u]; }
  const float mean = bsum(s, red) * (1.f / Cc);
  float q = 0.f;
#pragma unroll
  for (int u = 0; u < 4; ++u) { const float d = v[u] - mean; q += d * d; }
  const float rs = rsqrtf(bsum(q, red) * (1.f / Cc) + 1e-5f);
#pragma unroll
  for (int u = 0; u < 4; ++u) {
    const int c = tid * 4 + u;
    out[(size_t)tok * Cc + c] = (v[u] - mean) * rs * g[c] + b[c];
  }
}

__global__ __launch_bounds__(256) void k_mix6(
    const float* __restrict__ x, const float* __restrict__ g, const float* __restrict__ b,
    const float* __restrict__ mix, float* xr, float* xw, float* xk, float* xv,
    float* xa, float* xg_) {
  __shared__ float red[256];
  const int tok = blockIdx.x, tid = threadIdx.x;
  const int t = tok % Tt;
  const float* cur = x + (size_t)tok * Cc;
  const float* prv = cur - Cc;  // dereferenced only when t > 0
  float cv[4], pv[4]; float sc = 0.f, sp = 0.f;
#pragma unroll
  for (int u = 0; u < 4; ++u) {
    cv[u] = cur[tid * 4 + u]; sc += cv[u];
    pv[u] = (t > 0) ? prv[tid * 4 + u] : 0.f; sp += pv[u];
  }
  const float mc = bsum(sc, red) * (1.f / Cc);
  const float mp = bsum(sp, red) * (1.f / Cc);
  float qc = 0.f, qp = 0.f;
#pragma unroll
  for (int u = 0; u < 4; ++u) {
    const float d = cv[u] - mc; qc += d * d;
    const float e = pv[u] - mp; qp += e * e;
  }
  const float rc = rsqrtf(bsum(qc, red) * (1.f / Cc) + 1e-5f);
  const float rp = rsqrtf(bsum(qp, red) * (1.f / Cc) + 1e-5f);
#pragma unroll
  for (int u = 0; u < 4; ++u) {
    const int c = tid * 4 + u;
    const size_t o = (size_t)tok * Cc + c;
    const float xin = (cv[u] - mc) * rc * g[c] + b[c];
    const float pln = (t > 0) ? ((pv[u] - mp) * rp * g[c] + b[c]) : 0.f;
    const float xx = pln - xin;
    xr[o] = xin + xx * mix[0 * Cc + c]; xw[o]  = xin + xx * mix[1 * Cc + c];
    xk[o] = xin + xx * mix[2 * Cc + c]; xv[o]  = xin + xx * mix[3 * Cc + c];
    xa[o] = xin + xx * mix[4 * Cc + c]; xg_[o] = xin + xx * mix[5 * Cc + c];
  }
}

__global__ __launch_bounds__(256) void k_mix1(
    const float* __restrict__ x, const float* __restrict__ g, const float* __restrict__ b,
    const float* __restrict__ mix, float* __restrict__ out) {
  __shared__ float red[256];
  const int tok = blockIdx.x, tid = threadIdx.x;
  const int t = tok % Tt;
  const float* cur = x + (size_t)tok * Cc;
  const float* prv = cur - Cc;
  float cv[4], pv[4]; float sc = 0.f, sp = 0.f;
#pragma unroll
  for (int u = 0; u < 4; ++u) {
    cv[u] = cur[tid * 4 + u]; sc += cv[u];
    pv[u] = (t > 0) ? prv[tid * 4 + u] : 0.f; sp += pv[u];
  }
  const float mc = bsum(sc, red) * (1.f / Cc);
  const float mp = bsum(sp, red) * (1.f / Cc);
  float qc = 0.f, qp = 0.f;
#pragma unroll
  for (int u = 0; u < 4; ++u) {
    const float d = cv[u] - mc; qc += d * d;
    const float e = pv[u] - mp; qp += e * e;
  }
  const float rc = rsqrtf(bsum(qc, red) * (1.f / Cc) + 1e-5f);
  const float rp = rsqrtf(bsum(qp, red) * (1.f / Cc) + 1e-5f);
#pragma unroll
  for (int u = 0; u < 4; ++u) {
    const int c = tid * 4 + u;
    const size_t o = (size_t)tok * Cc + c;
    const float xc = (cv[u] - mc) * rc * g[c] + b[c];
    const float pl = (t > 0) ? ((pv[u] - mp) * rp * g[c] + b[c]) : 0.f;
    out[o] = xc + (pl - xc) * mix[c];
  }
}

__global__ void k_decay(const float* __restrict__ w0, const float* __restrict__ tmp,
                        float* __restrict__ w, int n) {
  const int i = blockIdx.x * blockDim.x + threadIdx.x;
  if (i >= n) return;
  const int c = i & (Cc - 1);
  const float z = -(w0[c] + tmp[i]);
  const float sp = (z > 20.f) ? z : log1pf(expf(z));
  w[i] = -sp - 0.5f;
}

__global__ void k_vblend(const float* __restrict__ v0, const float* __restrict__ tmp,
                         float* __restrict__ v, float* __restrict__ vfirst,
                         int layer0, int n) {
  const int i = blockIdx.x * blockDim.x + threadIdx.x;
  if (i >= n) return;
  if (layer0) { vfirst[i] = v[i]; return; }
  const int c = i & (Cc - 1);
  const float s = 1.f / (1.f + expf(-(v0[c] + tmp[i])));
  v[i] = v[i] + (vfirst[i] - v[i]) * s;
}

__global__ __launch_bounds__(256) void k_prep(
    const float* __restrict__ k, const float* __restrict__ a,
    const float* __restrict__ k_k, const float* __restrict__ k_a,
    float* __restrict__ k2, float* __restrict__ aa, float* __restrict__ bb) {
  __shared__ float hs[Hh];
  const int tok = blockIdx.x, tid = threadIdx.x;
  if (tid < Hh) hs[tid] = 0.f;
  __syncthreads();
  const int c0 = tid * 4, h = c0 >> 6;
  const size_t base = (size_t)tok * Cc;
  float kk[4]; float part = 0.f;
#pragma unroll
  for (int u = 0; u < 4; ++u) {
    const int c = c0 + u;
    kk[u] = k[base + c] * k_k[c];
    part += kk[u] * kk[u];
  }
  atomicAdd(&hs[h], part);
  __syncthreads();
  const float inv = 1.f / fmaxf(sqrtf(hs[h]), 1e-12f);
#pragma unroll
  for (int u = 0; u < 4; ++u) {
    const int c = c0 + u;
    const size_t o = base + c;
    const float kn = kk[u] * inv;
    const float av = a[o];
    aa[o] = -kn;
    bb[o] = kn * av;
    k2[o] = k[o] * (1.f + (av - 1.f) * k_a[c]);
  }
}

__global__ __launch_bounds__(64) void k_scan(
    const float* __restrict__ r, const float* __restrict__ w,
    const float* __restrict__ k2, const float* __restrict__ v,
    const float* __restrict__ aa, const float* __restrict__ bb,
    const float* __restrict__ ts, float* __restrict__ y) {
  const int bh = blockIdx.x, b = bh / Hh, h = bh % Hh;
  const int i = threadIdx.x;
  __shared__ float rs_[Nn], ws_[Nn], ks_[Nn], as_[Nn], bs_[Nn];
  float S[Nn];
  const float* t0 = ts + ((size_t)h * Nn + i) * Nn;
#pragma unroll
  for (int j = 0; j < Nn; ++j) S[j] = t0[j];
  for (int t = 0; t < Tt; ++t) {
    const size_t base = ((size_t)(b * Tt + t)) * Cc + h * Nn;
    rs_[i] = r[base + i];
    ws_[i] = expf(w[base + i]);
    ks_[i] = k2[base + i];
    as_[i] = aa[base + i];
    bs_[i] = bb[base + i];
    const float vi = v[base + i];
    __syncthreads();
    float sa = 0.f;
#pragma unroll
    for (int j = 0; j < Nn; ++j) sa += S[j] * as_[j];
    float yi = 0.f;
#pragma unroll
    for (int j = 0; j < Nn; ++j) {
      S[j] = S[j] * ws_[j] + sa * bs_[j] + vi * ks_[j];
      yi += S[j] * rs_[j];
    }
    y[base + i] = yi;
    __syncthreads();
  }
}

__global__ __launch_bounds__(256) void k_post(
    const float* __restrict__ y, const float* __restrict__ r,
    const float* __restrict__ k2, const float* __restrict__ v,
    const float* __restrict__ g, const float* __restrict__ lng,
    const float* __restrict__ lnb, const float* __restrict__ rk,
    float* __restrict__ z) {
  __shared__ float hs[Hh], hq[Hh], hd[Hh];
  const int tok = blockIdx.x, tid = threadIdx.x;
  if (tid < Hh) { hs[tid] = 0.f; hq[tid] = 0.f; hd[tid] = 0.f; }
  __syncthreads();
  const int c0 = tid * 4, h = c0 >> 6;
  const size_t base = (size_t)tok * Cc;
  float yv[4]; float s = 0.f, q = 0.f, d = 0.f;
#pragma unroll
  for (int u = 0; u < 4; ++u) {
    const size_t o = base + c0 + u;
    yv[u] = y[o];
    s += yv[u]; q += yv[u] * yv[u];
    d += r[o] * k2[o] * rk[c0 + u];
  }
  atomicAdd(&hs[h], s); atomicAdd(&hq[h], q); atomicAdd(&hd[h], d);
  __syncthreads();
  const float mean = hs[h] * (1.f / Nn);
  const float var = hq[h] * (1.f / Nn) - mean * mean;
  const float rsv = rsqrtf(var + 64e-5f);
  const float dot = hd[h];
#pragma unroll
  for (int u = 0; u < 4; ++u) {
    const int c = c0 + u;
    const size_t o = base + c;
    const float yn = (yv[u] - mean) * rsv * lng[c] + lnb[c];
    z[o] = (yn + dot * v[o]) * g[o];
  }
}

// ---------------------------------------------------------------------------
extern "C" void kernel_launch(void* const* d_in, const int* in_sizes, int n_in,
                              void* d_out, int out_size, void* d_ws, size_t ws_size,
                              hipStream_t stream) {
  (void)in_sizes; (void)n_in; (void)out_size; (void)ws_size;
  const int*   idx     = (const int*)d_in[0];
  const float* emb_w   = (const float*)d_in[1];
  const float* ln0_g   = (const float*)d_in[2];
  const float* ln0_b   = (const float*)d_in[3];
  const float* ln1_g   = (const float*)d_in[4];
  const float* ln1_b   = (const float*)d_in[5];
  const float* ln2_g   = (const float*)d_in[6];
  const float* ln2_b   = (const float*)d_in[7];
  const float* lnx_g   = (const float*)d_in[8];
  const float* lnx_b   = (const float*)d_in[9];
  const float* lnout_g = (const float*)d_in[10];
  const float* lnout_b = (const float*)d_in[11];
  const float* x_mix   = (const float*)d_in[12];
  const float* w0      = (const float*)d_in[13];
  const float* w1      = (const float*)d_in[14];
  const float* w2      = (const float*)d_in[15];
  const float* a0      = (const float*)d_in[16];
  const float* a1      = (const float*)d_in[17];
  const float* a2      = (const float*)d_in[18];
  const float* v0      = (const float*)d_in[19];
  const float* v1      = (const float*)d_in[20];
  const float* v2      = (const float*)d_in[21];
  const float* g1      = (const float*)d_in[22];
  const float* g2      = (const float*)d_in[23];
  const float* k_k     = (const float*)d_in[24];
  const float* k_a     = (const float*)d_in[25];
  const float* r_k     = (const float*)d_in[26];
  const float* Wr      = (const float*)d_in[27];
  const float* Wk      = (const float*)d_in[28];
  const float* Wv      = (const float*)d_in[29];
  const float* Wo      = (const float*)d_in[30];
  const float* ffn_xk  = (const float*)d_in[31];
  const float* ffn_k   = (const float*)d_in[32];
  const float* ffn_v   = (const float*)d_in[33];
  const float* tstate  = (const float*)d_in[34];
  const float* head_w  = (const float*)d_in[35];

  // ---- workspace: fp32 activations, then bf16 transposed weights ----
  float* W  = (float*)d_ws;
  const size_t BTC = (size_t)BT * Cc;
  auto Bf = [&](int i) { return W + (size_t)i * BTC; };
  float* H0 = W + 16 * BTC;                  // [BT, FF]
  float* L0 = H0 + (size_t)BT * FFf;         // [BT, 128] low-rank tmp
  __bf16* bw = (__bf16*)(L0 + (size_t)BT * 128);
  auto balloc = [&](size_t n) { __bf16* p = bw; bw += n; return p; };

  __bf16* tHead = balloc((size_t)Vv * Cc);
  __bf16 *tWr[Ll], *tWk[Ll], *tWv[Ll], *tWo[Ll];
  __bf16 *tw1[Ll], *tw2[Ll], *ta1[Ll], *ta2[Ll], *tv1[Ll], *tv2[Ll], *tg1[Ll], *tg2[Ll];
  __bf16 *tfk[Ll], *tfv[Ll];
  for (int i = 0; i < Ll; ++i) {
    tWr[i] = balloc((size_t)Cc * Cc); tWk[i] = balloc((size_t)Cc * Cc);
    tWv[i] = balloc((size_t)Cc * Cc); tWo[i] = balloc((size_t)Cc * Cc);
    tw1[i] = balloc((size_t)Cc * 64);  tw2[i] = balloc((size_t)64 * Cc);
    ta1[i] = balloc((size_t)Cc * 64);  ta2[i] = balloc((size_t)64 * Cc);
    tv1[i] = balloc((size_t)Cc * 32);  tv2[i] = balloc((size_t)32 * Cc);
    tg1[i] = balloc((size_t)Cc * 128); tg2[i] = balloc((size_t)128 * Cc);
    tfk[i] = balloc((size_t)Cc * FFf); tfv[i] = balloc((size_t)FFf * Cc);
  }

  auto cvt = [&](const float* src, __bf16* dst, int K, int N) {
    dim3 g((N + 31) / 32, (K + 31) / 32);
    k_cvt_t<<<g, 256, 0, stream>>>(src, dst, K, N);
  };
  cvt(head_w, tHead, Cc, Vv);
  for (int i = 0; i < Ll; ++i) {
    const size_t cc = (size_t)i * Cc * Cc;
    cvt(Wr + cc, tWr[i], Cc, Cc); cvt(Wk + cc, tWk[i], Cc, Cc);
    cvt(Wv + cc, tWv[i], Cc, Cc); cvt(Wo + cc, tWo[i], Cc, Cc);
    cvt(w1 + (size_t)i * Cc * 64,  tw1[i], Cc, 64);
    cvt(w2 + (size_t)i * 64 * Cc,  tw2[i], 64, Cc);
    cvt(a1 + (size_t)i * Cc * 64,  ta1[i], Cc, 64);
    cvt(a2 + (size_t)i * 64 * Cc,  ta2[i], 64, Cc);
    cvt(v1 + (size_t)i * Cc * 32,  tv1[i], Cc, 32);
    cvt(v2 + (size_t)i * 32 * Cc,  tv2[i], 32, Cc);
    cvt(g1 + (size_t)i * Cc * 128, tg1[i], Cc, 128);
    cvt(g2 + (size_t)i * 128 * Cc, tg2[i], 128, Cc);
    cvt(ffn_k + (size_t)i * Cc * FFf, tfk[i], Cc, FFf);
    cvt(ffn_v + (size_t)i * FFf * Cc, tfv[i], FFf, Cc);
  }

  // buffer roles:
  // 0:x 1:v_first 2:xr->k2 3:xw->aa 4:xk->bb 5:xv->z 6:xa->ffn_in 7:xg
  // 8:r 9:k 10:v 11:tmp 12:w(decay) 13:a 14:g 15:y
  dim3 blk(256);
  const int nEl = BT * Cc, gEl = (nEl + 255) / 256;

  k_embed_ln0<<<BT, blk, 0, stream>>>(idx, emb_w, ln0_g, ln0_b, Bf(0));

  for (int i = 0; i < Ll; ++i) {
    k_mix6<<<BT, blk, 0, stream>>>(Bf(0), ln1_g + i * Cc, ln1_b + i * Cc,
                                   x_mix + (size_t)i * 6 * Cc,
                                   Bf(2), Bf(3), Bf(4), Bf(5), Bf(6), Bf(7));
    gemm(Bf(2), tWr[i], Bf(8),  nullptr, BT, Cc, Cc, 0, stream);           // r
    gemm(Bf(4), tWk[i], Bf(9),  nullptr, BT, Cc, Cc, 0, stream);           // k
    gemm(Bf(5), tWv[i], Bf(10), nullptr, BT, Cc, Cc, 0, stream);           // v
    // decay path: tanh(xw@w1)@w2
    gemm(Bf(3), tw1[i], L0, nullptr, BT, 64, Cc, 1, stream);
    gemm(L0, tw2[i], Bf(11), nullptr, BT, Cc, 64, 0, stream);
    k_decay<<<gEl, blk, 0, stream>>>(w0 + i * Cc, Bf(11), Bf(12), nEl);
    // a path: sigmoid(a0 + (xa@a1)@a2)
    gemm(Bf(6), ta1[i], L0, nullptr, BT, 64, Cc, 0, stream);
    gemm(L0, ta2[i], Bf(13), a0 + i * Cc, BT, Cc, 64, 5, stream);
    // v residual-mix path
    if (i > 0) {
      gemm(Bf(5), tv1[i], L0, nullptr, BT, 32, Cc, 0, stream);
      gemm(L0, tv2[i], Bf(11), nullptr, BT, Cc, 32, 0, stream);
    }
    k_vblend<<<gEl, blk, 0, stream>>>(v0 + i * Cc, Bf(11), Bf(10), Bf(1),
                                      (i == 0) ? 1 : 0, nEl);
    // g path: sigmoid(xg@g1)@g2
    gemm(Bf(7), tg1[i], L0, nullptr, BT, 128, Cc, 2, stream);
    gemm(L0, tg2[i], Bf(14), nullptr, BT, Cc, 128, 0, stream);
    // state scan
    k_prep<<<BT, blk, 0, stream>>>(Bf(9), Bf(13), k_k + i * Cc, k_a + i * Cc,
                                   Bf(2), Bf(3), Bf(4));
    k_scan<<<Bb * Hh, 64, 0, stream>>>(Bf(8), Bf(12), Bf(2), Bf(10), Bf(3), Bf(4),
                                       tstate + (size_t)i * Hh * Nn * Nn, Bf(15));
    k_post<<<BT, blk, 0, stream>>>(Bf(15), Bf(8), Bf(2), Bf(10), Bf(14),
                                   lnx_g + i * Cc, lnx_b + i * Cc, r_k + i * Cc, Bf(5));
    gemm(Bf(5), tWo[i], Bf(0), Bf(0), BT, Cc, Cc, 4, stream);              // x += z@Wo
    // channel mix
    k_mix1<<<BT, blk, 0, stream>>>(Bf(0), ln2_g + i * Cc, ln2_b + i * Cc,
                                   ffn_xk + i * Cc, Bf(6));
    gemm(Bf(6), tfk[i], H0, nullptr, BT, FFf, Cc, 3, stream);
    gemm(H0, tfv[i], Bf(0), Bf(0), BT, Cc, FFf, 4, stream);
  }

  k_ln<<<BT, blk, 0, stream>>>(Bf(0), lnout_g, lnout_b, Bf(2));
  gemm(Bf(2), tHead, (float*)d_out, nullptr, BT, Vv, Cc, 0, stream);
}
